// SiameseDisplacementNet_30623116820527
// MI455X (gfx1250) — matrix operand order repaired and checked by reading, not compiled, not run on hardware
//
#include <hip/hip_runtime.h>

typedef __attribute__((ext_vector_type(16))) _Float16 v16h;
typedef __attribute__((ext_vector_type(8)))  _Float16 v8h;
typedef __attribute__((ext_vector_type(8)))  float    v8f;

// ---------------------------------------------------------------------------
// f32 -> f16 conversion (frames)
// ---------------------------------------------------------------------------
__global__ void to_half_kernel(const float* __restrict__ s, _Float16* __restrict__ d, int n) {
    int i = blockIdx.x * blockDim.x + threadIdx.x;
    if (i < n) d[i] = (_Float16)s[i];
}

// ---------------------------------------------------------------------------
// Implicit-GEMM convolution on WMMA f32 = f16 x f16 + f32.
// Block = 8 waves = CTB cout-tiles x PTB pixel-tiles of one 16x16 tile each.
// Per K-step (K chunk of 32):
//   - 256 threads cooperatively stage im2col B tiles [PTB][16pix][32K] (f16)
//     and converted weight A tiles [CTB][16row][32K] (f16) in LDS,
//   - each wave reads its fragments from LDS with wide ds loads laid out so
//     the ISA's 16-bit A/B per-lane K-packing is contiguous,
//   - v_wmma_f32_16x16x32_f16 accumulates in f32.
// Epilogue: +bias, optional leaky-relu, write f32 (pre-BN) or f16.
// ---------------------------------------------------------------------------
template<int CIN,int COUT,int KH,int KW,int STR,int PAD,
         int HIN,int WIN,int HOUT,int WOUT,bool FUSE_LRELU,bool OUTF16>
__global__ __launch_bounds__(256)
void conv_wmma_kernel(const _Float16* __restrict__ in, const float* __restrict__ wgt,
                      const float* __restrict__ bias,
                      float* __restrict__ out32, _Float16* __restrict__ out16) {
    constexpr int KHW    = KH * KW;
    constexpr int KTOT   = CIN * KHW;
    constexpr int KSTEPS = (KTOT + 31) / 32;
    constexpr int NPIX   = 32 * HOUT * WOUT;        // batch B = 32
    constexpr int CT     = COUT / 16;               // cout tiles (<= 8 for all layers)
    constexpr int CTB    = CT;                      // cout tiles per block
    constexpr int PTB    = 8 / CTB;                 // pixel tiles per block
    static_assert(8 % CTB == 0, "cout tiles must divide 8");
    static_assert((NPIX / 16) % PTB == 0, "exact grid");

    __shared__ _Float16 shB[PTB][16][32];           // [pixel tile][n][k]
    __shared__ _Float16 shA[CTB][16][32];           // [cout tile][m][k]

    const int wid  = threadIdx.x >> 5;
    const int lane = threadIdx.x & 31;
    const int ptl  = wid / CTB;                     // pixel tile within block
    const int ct   = wid % CTB;                     // cout tile within block

    const int pixBase = (blockIdx.x * PTB + ptl) * 16;   // 16 consecutive px, same row
    const int b   = pixBase / (HOUT * WOUT);
    const int rem = pixBase % (HOUT * WOUT);
    const int oy  = rem / WOUT;
    const int ox0 = rem % WOUT;

    const int hi = lane >> 4;                       // K-half select per ISA layout
    const int lm = lane & 15;                       // A: row M / B: col N

    v8f acc = {};
    for (int ks = 0; ks < KSTEPS; ++ks) {
        const int kbase = ks * 32;

        // ---- cooperative stage: im2col B tiles ----
        for (int t = threadIdx.x; t < PTB * 512; t += 256) {
            int p_l = t >> 9;
            int n   = (t >> 5) & 15;
            int k_o = t & 31;
            int k   = kbase + k_o;
            _Float16 v = (_Float16)0.0f;
            if (k < KTOT) {
                int pix = (blockIdx.x * PTB + p_l) * 16 + n;
                int bb  = pix / (HOUT * WOUT);
                int rr  = pix % (HOUT * WOUT);
                int oyb = rr / WOUT;
                int oxb = rr % WOUT;
                int cin = k / KHW;
                int r2  = k - cin * KHW;
                int ky  = r2 / KW;
                int kx  = r2 - ky * KW;
                int iy  = oyb * STR - PAD + ky;
                int ix  = oxb * STR - PAD + kx;
                if (iy >= 0 && iy < HIN && ix >= 0 && ix < WIN)
                    v = in[((bb * CIN + cin) * HIN + iy) * WIN + ix];
            }
            shB[p_l][n][k_o] = v;
        }
        // ---- cooperative stage: weights (f32 -> f16 once per block) ----
        for (int t = threadIdx.x; t < CTB * 512; t += 256) {
            int c_l = t >> 9;
            int m   = (t >> 5) & 15;
            int k_o = t & 31;
            int k   = kbase + k_o;
            float wv = (k < KTOT) ? wgt[(c_l * 16 + m) * KTOT + k] : 0.0f;
            shA[c_l][m][k_o] = (_Float16)wv;
            if (k + 32 < KTOT)
                __builtin_prefetch(&wgt[(c_l * 16 + m) * KTOT + k + 32], 0, 1);
        }
        __syncthreads();

        // ---- A fragment: lanes<16 take K {0..7,16..23}, lanes>=16 {8..15,24..31}
        v16h afrag;
        {
            const _Float16* ap = &shA[ct][lm][0];
            v8h a0 = *(const v8h*)(ap + hi * 8);          // K = hi*8 .. hi*8+7
            v8h a1 = *(const v8h*)(ap + 16 + hi * 8);     // K = 16+hi*8 ..
            #pragma unroll
            for (int e = 0; e < 8; ++e) { afrag[e] = a0[e]; afrag[e + 8] = a1[e]; }
        }
        // ---- B fragment: lanes<16 take K {0..7,16..23}, lanes>=16 {8..15,24..31}
        v16h bfrag = __builtin_shufflevector(*(const v8h*)&shB[ptl][lm][hi * 8], *(const v8h*)&shB[ptl][lm][16 + hi * 8], 0,1,2,3,4,5,6,7,8,9,10,11,12,13,14,15);

        acc = __builtin_amdgcn_wmma_f32_16x16x32_f16(
                  false, afrag, false, bfrag, (short)0, acc, false, false);
        __syncthreads();
    }

    // Epilogue: C/D 16x16 f32 layout — vgpr r, lanes<16 -> M=r, lanes>=16 -> M=r+8
    #pragma unroll
    for (int r = 0; r < 8; ++r) {
        int m  = r + hi * 8;
        int co = ct * 16 + m;
        float v = acc[r] + bias[co];
        if (FUSE_LRELU) v = (v > 0.0f) ? v : 0.1f * v;
        int idx = ((b * COUT + co) * HOUT + oy) * WOUT + ox0 + lm;
        if constexpr (OUTF16) out16[idx] = (_Float16)v;
        else                  out32[idx] = v;
    }
}

// ---------------------------------------------------------------------------
// BatchNorm statistics: one workgroup per channel, biased variance over (B,H,W)
// ---------------------------------------------------------------------------
template<int C,int HW>
__global__ __launch_bounds__(256)
void bn_stats_kernel(const float* __restrict__ x, float* __restrict__ mean,
                     float* __restrict__ invstd) {
    const int c = blockIdx.x;
    const int N = 32 * HW;
    float s = 0.0f, s2 = 0.0f;
    for (int i = threadIdx.x; i < N; i += blockDim.x) {
        int b = i / HW, r = i - b * HW;
        float v = x[(b * C + c) * HW + r];
        s += v; s2 += v * v;
    }
    __shared__ float sh[256], sh2[256];
    sh[threadIdx.x] = s; sh2[threadIdx.x] = s2;
    __syncthreads();
    for (int o = 128; o > 0; o >>= 1) {
        if (threadIdx.x < o) {
            sh[threadIdx.x]  += sh[threadIdx.x + o];
            sh2[threadIdx.x] += sh2[threadIdx.x + o];
        }
        __syncthreads();
    }
    if (threadIdx.x == 0) {
        float m   = sh[0] / (float)N;
        float var = sh2[0] / (float)N - m * m;
        mean[c]   = m;
        invstd[c] = rsqrtf(var + 1e-5f);
    }
}

// ---------------------------------------------------------------------------
// BN apply + leaky-relu; writes f16 (feeds next WMMA conv) or f32 (coarse)
// ---------------------------------------------------------------------------
template<int C,int HW,bool OUTF16>
__global__ void bn_apply_kernel(const float* __restrict__ x, const float* __restrict__ mean,
                                const float* __restrict__ invstd, const float* __restrict__ g,
                                const float* __restrict__ be,
                                float* __restrict__ o32, _Float16* __restrict__ o16) {
    int i = blockIdx.x * blockDim.x + threadIdx.x;
    constexpr int N = 32 * C * HW;
    if (i >= N) return;
    int c = (i / HW) % C;
    float v = (x[i] - mean[c]) * invstd[c] * g[c] + be[c];
    v = (v > 0.0f) ? v : 0.1f * v;
    if constexpr (OUTF16) o16[i] = (_Float16)v;
    else                  o32[i] = v;
}

// ---------------------------------------------------------------------------
// Per-pixel L2 normalization over 32 channels (in place, f32)
// ---------------------------------------------------------------------------
__global__ void l2norm_kernel(float* __restrict__ x) {  // (32,32,32,32)
    int p = blockIdx.x * blockDim.x + threadIdx.x;
    if (p >= 32 * 1024) return;
    int b = p >> 10, r = p & 1023;
    float vals[32];
    float s = 0.0f;
    #pragma unroll
    for (int c = 0; c < 32; ++c) { vals[c] = x[(b * 32 + c) * 1024 + r]; s += vals[c] * vals[c]; }
    float inv = 1.0f / fmaxf(sqrtf(s), 1e-12f);
    #pragma unroll
    for (int c = 0; c < 32; ++c) x[(b * 32 + c) * 1024 + r] = vals[c] * inv;
}

// ---------------------------------------------------------------------------
// Correlation: block per (b, oy). Stage 13 displaced rows of fB in LDS
// (13 x 32ch x 32w f32 = 53 KB), each thread owns one ox and ~22 displacements.
// Writes the 169-channel cost volume directly in f16 for the WMMA decoder.
// ---------------------------------------------------------------------------
__global__ __launch_bounds__(256)
void corr_kernel(const float* __restrict__ A, const float* __restrict__ Bf,
                 _Float16* __restrict__ cost) {
    const int b  = blockIdx.x >> 5;
    const int oy = blockIdx.x & 31;
    __shared__ float sB[13][32][32];               // [disp_row][channel][x]
    for (int t = threadIdx.x; t < 13 * 32 * 32; t += blockDim.x) {
        int i  = t >> 10;
        int c  = (t >> 5) & 31;
        int xx = t & 31;
        int hb = oy + i - 6;
        float v = 0.0f;
        if (hb >= 0 && hb < 32) v = Bf[(b * 32 + c) * 1024 + hb * 32 + xx];
        sB[i][c][xx] = v;
    }
    __syncthreads();

    const int ox  = threadIdx.x & 31;
    const int grp = threadIdx.x >> 5;              // 8 groups cover 169 displacements
    float av[32];
    #pragma unroll
    for (int c = 0; c < 32; ++c) av[c] = A[(b * 32 + c) * 1024 + oy * 32 + ox];

    for (int d = grp; d < 169; d += 8) {
        int i = d / 13, j = d - i * 13;
        int wb = ox + j - 6;
        float s = 0.0f;
        if (wb >= 0 && wb < 32) {
            #pragma unroll
            for (int c = 0; c < 32; ++c) s += av[c] * sB[i][c][wb];
        }
        cost[((b * 169 + d) * 32 + oy) * 32 + ox] = (_Float16)s;
    }
}

// ---------------------------------------------------------------------------
// 2x nearest upsample of x2 (64ch,32x32) + concat skips -> (96ch,64x64) f16
// ---------------------------------------------------------------------------
__global__ void upcat_kernel(const _Float16* __restrict__ x2, const _Float16* __restrict__ fA,
                             const _Float16* __restrict__ fB, _Float16* __restrict__ cat) {
    int i = blockIdx.x * blockDim.x + threadIdx.x;
    const int N = 32 * 96 * 4096;
    if (i >= N) return;
    int xx = i & 63;
    int yy = (i >> 6) & 63;
    int ch = (i >> 12) % 96;
    int b  = i / (96 * 4096);
    _Float16 v;
    if (ch < 64)      v = x2[((b * 64 + ch)       * 32 + (yy >> 1)) * 32 + (xx >> 1)];
    else if (ch < 80) v = fA[((b * 16 + (ch - 64)) * 64 + yy) * 64 + xx];
    else              v = fB[((b * 16 + (ch - 80)) * 64 + yy) * 64 + xx];
    cat[i] = v;
}

// ---------------------------------------------------------------------------
// Final 1x1 conv 32 -> 2, f32 output
// ---------------------------------------------------------------------------
__global__ void convo_kernel(const _Float16* __restrict__ y2, const float* __restrict__ wo,
                             const float* __restrict__ bo, float* __restrict__ out) {
    int p = blockIdx.x * blockDim.x + threadIdx.x;   // (b,y,x)
    if (p >= 32 * 4096) return;
    int b = p >> 12, r = p & 4095;
    float s0 = bo[0], s1 = bo[1];
    #pragma unroll
    for (int c = 0; c < 32; ++c) {
        float v = (float)y2[(b * 32 + c) * 4096 + r];
        s0 += wo[c] * v;
        s1 += wo[32 + c] * v;
    }
    out[(b * 2 + 0) * 4096 + r] = s0;
    out[(b * 2 + 1) * 4096 + r] = s1;
}

// ---------------------------------------------------------------------------
extern "C" void kernel_launch(void* const* d_in, const int* in_sizes, int n_in,
                              void* d_out, int out_size, void* d_ws, size_t ws_size,
                              hipStream_t stream) {
    (void)in_sizes; (void)n_in; (void)out_size; (void)ws_size;

    const float* frameA = (const float*)d_in[0];
    const float* frameB = (const float*)d_in[1];
    const float* w1  = (const float*)d_in[2];  const float* b1  = (const float*)d_in[3];
    const float* g1  = (const float*)d_in[4];  const float* be1 = (const float*)d_in[5];
    const float* w2  = (const float*)d_in[6];  const float* b2  = (const float*)d_in[7];
    const float* g2  = (const float*)d_in[8];  const float* be2 = (const float*)d_in[9];
    const float* w3  = (const float*)d_in[10]; const float* b3  = (const float*)d_in[11];
    const float* g3  = (const float*)d_in[12]; const float* be3 = (const float*)d_in[13];
    const float* wc1 = (const float*)d_in[14]; const float* bc1 = (const float*)d_in[15];
    const float* wc2 = (const float*)d_in[16]; const float* bc2 = (const float*)d_in[17];
    const float* wf1 = (const float*)d_in[18]; const float* bf1 = (const float*)d_in[19];
    const float* wf2 = (const float*)d_in[20]; const float* bf2 = (const float*)d_in[21];
    const float* wo  = (const float*)d_in[22]; const float* bo  = (const float*)d_in[23];
    float* out = (float*)d_out;

    char* ws = (char*)d_ws;
    size_t off = 0;
    auto alloc = [&](size_t bytes) -> char* {
        char* p = ws + off;
        off += (bytes + 255) & ~(size_t)255;
        return p;
    };

    _Float16* fAh     = (_Float16*)alloc(131072u * 2);          // frameA f16
    _Float16* fBh     = (_Float16*)alloc(131072u * 2);          // frameB f16
    float*    rawA    = (float*)   alloc(2097152u * 4);         // encoder scratch (frame A)
    float*    rawB    = (float*)   alloc(2097152u * 4);         // encoder scratch (frame B)
    _Float16* fineAh  = (_Float16*)alloc(2097152u * 2);         // (32,16,64,64) f16
    _Float16* fineBh  = (_Float16*)alloc(2097152u * 2);
    _Float16* hAh     = (_Float16*)alloc(1048576u * 2);         // (32,32,32,32) f16
    _Float16* hBh     = (_Float16*)alloc(1048576u * 2);
    float*    coarseA = (float*)   alloc(1048576u * 4);         // (32,32,32,32) f32
    float*    coarseB = (float*)   alloc(1048576u * 4);
    _Float16* cost    = (_Float16*)alloc(5537792u * 2);         // (32,169,32,32) f16
    _Float16* x1      = (_Float16*)alloc(4194304u * 2);         // (32,128,32,32)
    _Float16* x2      = (_Float16*)alloc(2097152u * 2);         // (32,64,32,32)
    _Float16* cat     = (_Float16*)alloc(12582912u * 2);        // (32,96,64,64)
    _Float16* y1      = (_Float16*)alloc(8388608u * 2);         // (32,64,64,64)
    _Float16* y2      = (_Float16*)alloc(4194304u * 2);         // (32,32,64,64)
    float*    meanb   = (float*)   alloc(128u * 4);
    float*    invb    = (float*)   alloc(128u * 4);

    to_half_kernel<<<512, 256, 0, stream>>>(frameA, fAh, 131072);
    to_half_kernel<<<512, 256, 0, stream>>>(frameB, fBh, 131072);

    // ---- encoder, frame A ----
    conv_wmma_kernel<1,16,7,7,1,3,64,64,64,64,false,false><<<1024,256,0,stream>>>(fAh, w1, b1, rawA, nullptr);
    bn_stats_kernel<16,4096><<<16,256,0,stream>>>(rawA, meanb, invb);
    bn_apply_kernel<16,4096,true><<<8192,256,0,stream>>>(rawA, meanb, invb, g1, be1, nullptr, fineAh);
    conv_wmma_kernel<16,32,5,5,2,2,64,64,32,32,false,false><<<512,256,0,stream>>>(fineAh, w2, b2, rawA, nullptr);
    bn_stats_kernel<32,1024><<<32,256,0,stream>>>(rawA, meanb, invb);
    bn_apply_kernel<32,1024,true><<<4096,256,0,stream>>>(rawA, meanb, invb, g2, be2, nullptr, hAh);
    conv_wmma_kernel<32,32,3,3,1,1,32,32,32,32,false,false><<<512,256,0,stream>>>(hAh, w3, b3, rawA, nullptr);
    bn_stats_kernel<32,1024><<<32,256,0,stream>>>(rawA, meanb, invb);
    bn_apply_kernel<32,1024,false><<<4096,256,0,stream>>>(rawA, meanb, invb, g3, be3, coarseA, nullptr);
    l2norm_kernel<<<128,256,0,stream>>>(coarseA);

    // ---- encoder, frame B ----
    conv_wmma_kernel<1,16,7,7,1,3,64,64,64,64,false,false><<<1024,256,0,stream>>>(fBh, w1, b1, rawB, nullptr);
    bn_stats_kernel<16,4096><<<16,256,0,stream>>>(rawB, meanb, invb);
    bn_apply_kernel<16,4096,true><<<8192,256,0,stream>>>(rawB, meanb, invb, g1, be1, nullptr, fineBh);
    conv_wmma_kernel<16,32,5,5,2,2,64,64,32,32,false,false><<<512,256,0,stream>>>(fineBh, w2, b2, rawB, nullptr);
    bn_stats_kernel<32,1024><<<32,256,0,stream>>>(rawB, meanb, invb);
    bn_apply_kernel<32,1024,true><<<4096,256,0,stream>>>(rawB, meanb, invb, g2, be2, nullptr, hBh);
    conv_wmma_kernel<32,32,3,3,1,1,32,32,32,32,false,false><<<512,256,0,stream>>>(hBh, w3, b3, rawB, nullptr);
    bn_stats_kernel<32,1024><<<32,256,0,stream>>>(rawB, meanb, invb);
    bn_apply_kernel<32,1024,false><<<4096,256,0,stream>>>(rawB, meanb, invb, g3, be3, coarseB, nullptr);
    l2norm_kernel<<<128,256,0,stream>>>(coarseB);

    // ---- correlation cost volume ----
    corr_kernel<<<1024,256,0,stream>>>(coarseA, coarseB, cost);

    // ---- decoder ----
    conv_wmma_kernel<169,128,3,3,1,1,32,32,32,32,true,true><<<2048,256,0,stream>>>(cost, wc1, bc1, nullptr, x1);
    conv_wmma_kernel<128,64,3,3,1,1,32,32,32,32,true,true><<<1024,256,0,stream>>>(x1, wc2, bc2, nullptr, x2);
    upcat_kernel<<<49152,256,0,stream>>>(x2, fineAh, fineBh, cat);
    conv_wmma_kernel<96,64,3,3,1,1,64,64,64,64,true,true><<<4096,256,0,stream>>>(cat, wf1, bf1, nullptr, y1);
    conv_wmma_kernel<64,32,3,3,1,1,64,64,64,64,true,true><<<2048,256,0,stream>>>(y1, wf2, bf2, nullptr, y2);
    convo_kernel<<<512,256,0,stream>>>(y2, wo, bo, out);
}
